// Wav2vec2AudioFrameClassRNNHead_71708773974133
// MI455X (gfx1250) — matrix operand order repaired and checked by reading, not compiled, not run on hardware
//
#include <hip/hip_runtime.h>
#include <math.h>

typedef __bf16 bf16;
typedef __attribute__((ext_vector_type(16))) __bf16 bf16x16;
typedef __attribute__((ext_vector_type(8)))  __bf16 bf16x8;
typedef __attribute__((ext_vector_type(8)))  float  floatx8;

#define B_     32
#define T_     1024
#define D_     768
#define H_     768
#define G3_    2304   // 3*H
#define INNER_ 512
#define NOUT_  39
#define NPADO_ 48

// ---------------------------------------------------------------------------
// WMMA fragment loads straight from row-major bf16 memory, per CDNA5 ISA
// VGPR layouts (CDNA5 ISA §7.12.2).
// A: 16x32 (MxK). lane&15 = M row; kh = (lane>>4)*8;
//    elems[0..7] = K(kh..kh+7), elems[8..15] = K(16+kh..16+kh+7).
// B: 32x16 (KxN) with B[k][n] = W[n][k] (W row-major [N,K]):
//    lane&15 = column n (a row of W); kh = (lane>>4)*8;
//    elems[0..7] = K(kh..kh+7), elems[8..15] = K(16+kh..16+kh+7).
// ---------------------------------------------------------------------------
__device__ __forceinline__ bf16x16 ldA(const bf16* __restrict__ base, int ld, int lane) {
  const int m  = lane & 15;
  const int kh = (lane >> 4) << 3;
  const bf16* q = base + (size_t)m * ld + kh;
  bf16x8 lo = *reinterpret_cast<const bf16x8*>(q);
  bf16x8 hi = *reinterpret_cast<const bf16x8*>(q + 16);
  return __builtin_shufflevector(lo, hi, 0,1,2,3,4,5,6,7,8,9,10,11,12,13,14,15);
}

__device__ __forceinline__ bf16x16 ldB(const bf16* __restrict__ wbase, int ld, int lane) {
  const int n  = lane & 15;
  const int kh = (lane >> 4) << 3;
  const bf16* q = wbase + (size_t)n * ld + kh;
  bf16x8 lo = *reinterpret_cast<const bf16x8*>(q);
  bf16x8 hi = *reinterpret_cast<const bf16x8*>(q + 16);
  return __builtin_shufflevector(lo, hi, 0,1,2,3,4,5,6,7,8,9,10,11,12,13,14,15);
}

__device__ __forceinline__ floatx8 wmma_bf16(bf16x16 a, bf16x16 b, floatx8 c) {
  return __builtin_amdgcn_wmma_f32_16x16x32_bf16(false, a, false, b, (short)0, c,
                                                 false, false);
}

// ---------------------------------------------------------------------------
// Tiled WMMA GEMM: out[M,N] = act(A[M,K] @ W[N,K]^T + bias)
// Block = 256 threads = 8 waves; each wave owns one 16-row M tile and NT
// N-tiles (A-fragment register reuse). Fully branch-free hot loop with
// register double-buffering (prefetch next k-step during current WMMAs).
// ACT: 0 = bias -> bf16 (ld=Npad), 1 = SELU -> bf16, 2 = tanh -> f32 (ld=Nreal)
// ---------------------------------------------------------------------------
template <int ACT, int NT>
__global__ __launch_bounds__(256) void gemm_wmma(
    const bf16* __restrict__ A, const bf16* __restrict__ W,
    const float* __restrict__ bias, void* __restrict__ out,
    int Npad, int K, int Nreal)
{
  const int lane  = threadIdx.x & 31;
  const int wave  = threadIdx.x >> 5;
  const int m0    = blockIdx.y * 128 + wave * 16;
  const int nbase = blockIdx.x * (NT * 16);

  const bf16* Ab = A + (size_t)m0 * K;
  const bf16* Wb[NT];
#pragma unroll
  for (int nn = 0; nn < NT; ++nn) Wb[nn] = W + (size_t)(nbase + nn * 16) * K;

  floatx8 acc[NT] = {};

  // preload k = 0
  bf16x16 a = ldA(Ab, K, lane);
  bf16x16 b[NT];
#pragma unroll
  for (int nn = 0; nn < NT; ++nn) b[nn] = ldB(Wb[nn], K, lane);

  for (int k0 = 32; k0 < K; k0 += 32) {
    bf16x16 an = ldA(Ab + k0, K, lane);
    bf16x16 bn[NT];
#pragma unroll
    for (int nn = 0; nn < NT; ++nn) bn[nn] = ldB(Wb[nn] + k0, K, lane);
#pragma unroll
    for (int nn = 0; nn < NT; ++nn) acc[nn] = wmma_bf16(a, b[nn], acc[nn]);
    a = an;
#pragma unroll
    for (int nn = 0; nn < NT; ++nn) b[nn] = bn[nn];
  }
#pragma unroll
  for (int nn = 0; nn < NT; ++nn) acc[nn] = wmma_bf16(a, b[nn], acc[nn]);

  // epilogue
  const int cn = lane & 15;
  const int mo = (lane >> 4) << 3;
#pragma unroll
  for (int nn = 0; nn < NT; ++nn) {
    const int col = nbase + nn * 16 + cn;
    const float bv = (ACT == 2) ? (col < Nreal ? bias[col] : 0.f) : bias[col];
#pragma unroll
    for (int r = 0; r < 8; ++r) {
      const int row = m0 + mo + r;
      float v = acc[nn][r] + bv;
      if (ACT == 1) {
        const float kScale = 1.0507009873554805f, kAlpha = 1.6732632423543772f;
        v = kScale * (v > 0.f ? v : kAlpha * (__expf(v) - 1.f));
        reinterpret_cast<bf16*>(out)[(size_t)row * Npad + col] = (bf16)v;
      } else if (ACT == 2) {
        if (col < Nreal)
          reinterpret_cast<float*>(out)[(size_t)row * Nreal + col] = tanhf(v);
      } else {
        reinterpret_cast<bf16*>(out)[(size_t)row * Npad + col] = (bf16)v;
      }
    }
  }
}

// ---------------------------------------------------------------------------
// Device-wide sense barrier for the persistent recurrent kernel.
// bar[0] = arrival count, bar[16] = monotonically-increasing generation.
// ---------------------------------------------------------------------------
__device__ __forceinline__ void grid_sync(unsigned* bar) {
  __threadfence();
  __syncthreads();
  if (threadIdx.x == 0) {
    unsigned* cnt = bar;
    unsigned* gen = bar + 16;
    const unsigned g = __atomic_load_n(gen, __ATOMIC_ACQUIRE);
    const unsigned arrived = atomicAdd(cnt, 1u);
    if (arrived == gridDim.x - 1u) {
      atomicExch(cnt, 0u);
      __threadfence();
      atomicAdd(gen, 1u);
    } else {
      while (__atomic_load_n(gen, __ATOMIC_ACQUIRE) == g)
        __builtin_amdgcn_s_sleep(2);
    }
  }
  __syncthreads();
}

// ---------------------------------------------------------------------------
// Persistent GRU recurrence for one layer.
// Per step: stage h[32,768] bf16 (48 KB) into LDS with coalesced b128 loads,
// gh = h @ w_hh^T + b_hh via WMMA (A-fragments served from LDS, 288 tiles over
// 256 waves), grid barrier, fused PyTorch gate math in f32, grid barrier.
// ---------------------------------------------------------------------------
__global__ __launch_bounds__(256) void gru_recurrent(
    const bf16* __restrict__ xproj,   // [B,T,3H] bf16, b_ih already added
    const bf16* __restrict__ whh,     // [3H,H] bf16
    const float* __restrict__ bhh,    // [3H]
    bf16* __restrict__ ys,            // [B,T,H] bf16 out
    float* __restrict__ gh,           // [B,3H] scratch
    bf16* __restrict__ h,             // [B,H] state
    unsigned* __restrict__ bar, int T)
{
  __shared__ bf16 hsh[B_ * H_];                  // 48 KB

  const int lane     = threadIdx.x & 31;
  const int gtid     = blockIdx.x * blockDim.x + threadIdx.x;
  const int nthreads = gridDim.x * blockDim.x;   // 32 * 256 = 8192
  const int gwave    = gtid >> 5;
  const int nwaves   = nthreads >> 5;            // 256
  const int ntiles   = 2 * (G3_ / 16);           // 288

  for (int e = gtid; e < B_ * H_; e += nthreads) h[e] = (bf16)0.f;
  grid_sync(bar);

  for (int t = 0; t < T; ++t) {
    // --- stage h into LDS (coalesced 16B chunks) --------------------------
    {
      const uint4* src = reinterpret_cast<const uint4*>(h);
      uint4* dst = reinterpret_cast<uint4*>(hsh);
      constexpr int NV = (B_ * H_ * 2) / 16;     // 3072 vectors
#pragma unroll
      for (int i = 0; i < NV / 256; ++i)
        dst[i * 256 + threadIdx.x] = src[i * 256 + threadIdx.x];
    }
    __syncthreads();

    // --- gh = h @ whh^T + bhh (A from LDS, B from L2-resident weights) ----
    for (int tile = gwave; tile < ntiles; tile += nwaves) {   // wave-uniform
      const int m0 = (tile & 1) << 4;
      const int n0 = (tile >> 1) << 4;
      const bf16* Wb = whh + (size_t)n0 * H_;
      floatx8 acc = {};
      bf16x16 a = ldA(hsh + m0 * H_, H_, lane);
      bf16x16 b = ldB(Wb, H_, lane);
      for (int k0 = 32; k0 < H_; k0 += 32) {
        bf16x16 an = ldA(hsh + m0 * H_ + k0, H_, lane);
        bf16x16 bn = ldB(Wb + k0, H_, lane);
        acc = wmma_bf16(a, b, acc);
        a = an; b = bn;
      }
      acc = wmma_bf16(a, b, acc);

      const int cn = lane & 15;
      const int mo = (lane >> 4) << 3;
      const int col = n0 + cn;
      const float bv = bhh[col];
#pragma unroll
      for (int r = 0; r < 8; ++r)
        gh[(m0 + mo + r) * G3_ + col] = acc[r] + bv;
    }
    grid_sync(bar);

    // --- fused gate math: r,z,n gates + state update ----------------------
    for (int e = gtid; e < B_ * H_; e += nthreads) {
      const int b = e / H_;
      const int j = e - b * H_;
      const size_t xo = ((size_t)b * T + t) * G3_ + j;
      const float gir = (float)xproj[xo];
      const float giz = (float)xproj[xo + H_];
      const float gin = (float)xproj[xo + 2 * H_];
      const float* g = gh + (size_t)b * G3_ + j;
      const float ghr = g[0], ghz = g[H_], ghn = g[2 * H_];
      const float hp = (float)h[e];
      const float r = 1.f / (1.f + __expf(-(gir + ghr)));
      const float z = 1.f / (1.f + __expf(-(giz + ghz)));
      const float n = tanhf(gin + r * ghn);
      const float hn = (1.f - z) * n + z * hp;
      h[e] = (bf16)hn;
      ys[((size_t)b * T + t) * H_ + j] = (bf16)hn;
    }
    grid_sync(bar);
  }
}

// ---------------------------------------------------------------------------
__global__ void cast_f32_bf16(const float* __restrict__ src,
                              bf16* __restrict__ dst, int n) {
  int i = blockIdx.x * blockDim.x + threadIdx.x;
  const int stride = gridDim.x * blockDim.x;
  for (; i < n; i += stride) dst[i] = (bf16)src[i];
}

__global__ void pack_outw(const float* __restrict__ src, bf16* __restrict__ dst) {
  const int i = blockIdx.x * blockDim.x + threadIdx.x;   // over 48*512
  if (i >= NPADO_ * INNER_) return;
  const int n = i / INNER_;
  const int k = i - n * INNER_;
  dst[i] = (n < NOUT_) ? (bf16)src[n * INNER_ + k] : (bf16)0.f;
}

// ---------------------------------------------------------------------------
extern "C" void kernel_launch(void* const* d_in, const int* in_sizes, int n_in,
                              void* d_out, int out_size, void* d_ws, size_t ws_size,
                              hipStream_t stream) {
  const float* features = (const float*)d_in[0];
  // d_in[1] = padding_mask (unused by forward math)
  const float* w_ih0 = (const float*)d_in[2];
  const float* w_hh0 = (const float*)d_in[3];
  const float* b_ih0 = (const float*)d_in[4];
  const float* b_hh0 = (const float*)d_in[5];
  const float* w_ih1 = (const float*)d_in[6];
  const float* w_hh1 = (const float*)d_in[7];
  const float* b_ih1 = (const float*)d_in[8];
  const float* b_hh1 = (const float*)d_in[9];
  const float* fc_w  = (const float*)d_in[10];
  const float* fc_b  = (const float*)d_in[11];
  const float* out_w = (const float*)d_in[12];
  const float* out_b = (const float*)d_in[13];
  float* out = (float*)d_out;

  char* ws = (char*)d_ws;
  size_t off = 0;
  auto alloc = [&](size_t bytes) -> char* {
    char* p = ws + off;
    off = (off + bytes + 255) & ~(size_t)255;
    return p;
  };
  unsigned* bar   = (unsigned*)alloc(256);
  bf16* featb = (bf16*)alloc((size_t)B_ * T_ * D_ * 2);
  bf16* ys0   = (bf16*)alloc((size_t)B_ * T_ * H_ * 2);
  bf16* ys1   = (bf16*)alloc((size_t)B_ * T_ * H_ * 2);
  bf16* xproj = (bf16*)alloc((size_t)B_ * T_ * G3_ * 2);
  bf16* fcb   = (bf16*)alloc((size_t)B_ * T_ * INNER_ * 2);
  bf16* wih0b = (bf16*)alloc((size_t)G3_ * D_ * 2);
  bf16* whh0b = (bf16*)alloc((size_t)G3_ * H_ * 2);
  bf16* wih1b = (bf16*)alloc((size_t)G3_ * H_ * 2);
  bf16* whh1b = (bf16*)alloc((size_t)G3_ * H_ * 2);
  bf16* fcwb  = (bf16*)alloc((size_t)INNER_ * H_ * 2);
  bf16* outwb = (bf16*)alloc((size_t)NPADO_ * INNER_ * 2);
  float* gh   = (float*)alloc((size_t)B_ * G3_ * 4);
  bf16* hbuf  = (bf16*)alloc((size_t)B_ * H_ * 2);
  if (off > ws_size) return;   // workspace too small: bail deterministically

  hipMemsetAsync(bar, 0, 256, stream);

  cast_f32_bf16<<<4096, 256, 0, stream>>>(features, featb, B_ * T_ * D_);
  cast_f32_bf16<<<2048, 256, 0, stream>>>(w_ih0, wih0b, G3_ * D_);
  cast_f32_bf16<<<2048, 256, 0, stream>>>(w_hh0, whh0b, G3_ * H_);
  cast_f32_bf16<<<2048, 256, 0, stream>>>(w_ih1, wih1b, G3_ * H_);
  cast_f32_bf16<<<2048, 256, 0, stream>>>(w_hh1, whh1b, G3_ * H_);
  cast_f32_bf16<<<512,  256, 0, stream>>>(fc_w, fcwb, INNER_ * H_);
  pack_outw<<<(NPADO_ * INNER_ + 255) / 256, 256, 0, stream>>>(out_w, outwb);

  const int M = B_ * T_;                        // 32768 rows
  dim3 blk(256);

  // Layer 0: input projection + recurrence
  gemm_wmma<0, 4><<<dim3(G3_ / 64, M / 128), blk, 0, stream>>>(
      featb, wih0b, b_ih0, xproj, G3_, D_, G3_);
  gru_recurrent<<<32, 256, 0, stream>>>(xproj, whh0b, b_hh0, ys0, gh, hbuf, bar, T_);

  // Layer 1: input projection + recurrence (xproj buffer reused)
  gemm_wmma<0, 4><<<dim3(G3_ / 64, M / 128), blk, 0, stream>>>(
      ys0, wih1b, b_ih1, xproj, G3_, H_, G3_);
  gru_recurrent<<<32, 256, 0, stream>>>(xproj, whh1b, b_hh1, ys1, gh, hbuf, bar, T_);

  // fc + SELU
  gemm_wmma<1, 4><<<dim3(INNER_ / 64, M / 128), blk, 0, stream>>>(
      ys1, fcwb, fc_b, fcb, INNER_, H_, INNER_);

  // out_proj + tanh (N padded 39 -> 48 = exactly 3 tiles, f32 output)
  gemm_wmma<2, 3><<<dim3(1, M / 128), blk, 0, stream>>>(
      fcb, outwb, out_b, out, NPADO_, INNER_, NOUT_);
}
